// VectorQuantizer_73057393704983
// MI455X (gfx1250) — matrix operand order repaired and checked by reading, not compiled, not run on hardware
//
#include <hip/hip_runtime.h>
#include <hip/hip_bf16.h>
#include <math.h>

typedef __attribute__((ext_vector_type(16))) _Float16 v16h;
typedef __attribute__((ext_vector_type(8)))  float    v8f;
typedef __attribute__((ext_vector_type(2)))  _Float16 h2;
typedef __attribute__((ext_vector_type(4)))  int      v4i;

#define N_ROWS 32768              // 32 * 32 * 32 pixels
#define C_DIM  512
#define K_CODES 1024
#define NQ (32 * 512 * 32 * 32)   // 16777216 quantized elements
#define CH_HALVES 32768           // one B chunk: 4 n-tiles * 16 kt * 512 halves (64 KB)

union PackH { h2 h; unsigned u; };

#if defined(__has_builtin)
#if __has_builtin(__builtin_amdgcn_global_load_async_to_lds_b128) && \
    __has_builtin(__builtin_amdgcn_s_wait_asynccnt)
#define HAVE_ASYNC_LDS 1
#endif
#endif

#if defined(HAVE_ASYNC_LDS)
#define WAIT_ASYNC(n) __builtin_amdgcn_s_wait_asynccnt(n)
typedef __attribute__((address_space(1))) v4i g_v4i;   // global (AS1 / __device__)
typedef __attribute__((address_space(3))) v4i s_v4i;   // LDS    (AS3 / __shared__)
#else
#define WAIT_ASYNC(n)
#endif

// copy one 64KB B chunk (32768 halves) global -> LDS with 256 threads
__device__ __forceinline__ void copy_chunk(const _Float16* __restrict__ src,
                                           _Float16* dst, int tid) {
#pragma unroll
    for (int p = 0; p < 16; ++p) {
        int off = (p * 256 + tid) * 8;      // halves, 16B per thread per pass
#if defined(HAVE_ASYNC_LDS)
        __builtin_amdgcn_global_load_async_to_lds_b128(
            (g_v4i*)(src + off),
            (s_v4i*)(dst + off), 0, 0);
#else
        uint4 v = *(const uint4*)(src + off);
        *(uint4*)(dst + off) = v;
#endif
    }
}

// ---------------------------------------------------------------------------
// Kernel 1: x [B,C,W,H] -> xs = sigmoid(100x)*2-1 (f16) in WMMA A-frag order.
// One block per 16-row m-tile: coalesced x reads (contiguous along H) into
// LDS, then fragment-swizzled, coalesced packed stores.
// A-frag (16-bit A 16x32): lane L (row m=L%16, g=L/16), VGPR v:
//   v<4 -> K = 2v+8g ; v>=4 -> K = 16+2(v-4)+8g
// ---------------------------------------------------------------------------
__global__ void __launch_bounds__(256) vq_pack_a(const float* __restrict__ x,
                                                 unsigned* __restrict__ A) {
    __shared__ _Float16 xsh[C_DIM * 16];    // 16 KB: [c][m_local]
    const int mt  = blockIdx.x;             // 0..2047
    const int tid = threadIdx.x;
    const int b   = mt >> 6;
    const int w   = (mt >> 1) & 31;
    const int h0  = (mt & 1) << 4;
    const float* xb = x + (b * 512) * 1024 + w * 32 + h0;
#pragma unroll 4
    for (int p = 0; p < 32; ++p) {
        int li = p * 256 + tid;             // 0..8191 : c*16 + h
        int c = li >> 4, hh = li & 15;
        float xv = xb[c * 1024 + hh];       // coalesced: 64B per channel
        float s  = 1.0f / (1.0f + __expf(-100.0f * xv));
        xsh[c * 16 + hh] = (_Float16)(s * 2.0f - 1.0f);
    }
    __syncthreads();
    unsigned* Ab = A + mt * 4096;           // 16 kt frags * 256 uints
#pragma unroll 4
    for (int j = 0; j < 16; ++j) {
        int u = j * 256 + tid;              // coalesced store index
        int kt = u >> 8, r = u & 255, lane = r >> 3, v = r & 7;
        int g = lane >> 4;
        int kin = (v < 4) ? (2 * v + 8 * g) : (16 + 2 * (v - 4) + 8 * g);
        int k = kt * 32 + kin;
        int ml = lane & 15;
        PackH p2;
        p2.h[0] = xsh[k * 16 + ml];
        p2.h[1] = xsh[(k + 1) * 16 + ml];
        Ab[u] = p2.u;
    }
}

// ---------------------------------------------------------------------------
// Kernel 2: codebook [1024,512] f32 -> f16 B-fragments (32x16, B[k][n]=cb[n][k])
// Lane L: n = L%16, g = L/16; VGPR v holds K = (v<4 ? 8g+2v : 16+8g+2(v-4)), +1
// ---------------------------------------------------------------------------
__global__ void __launch_bounds__(256) vq_pack_b(const float* __restrict__ cb,
                                                 unsigned* __restrict__ B) {
    unsigned t    = blockIdx.x * 256u + threadIdx.x;   // 262,144 threads
    unsigned fb   = t >> 8;
    unsigned r    = t & 255u;
    unsigned lane = r >> 3;
    unsigned e    = r & 7u;
    unsigned nt   = fb >> 4;
    unsigned kt   = fb & 15u;
    unsigned n    = nt * 16u + (lane & 15u);
    unsigned g    = lane >> 4;
    unsigned k    = kt * 32u + ((e < 4u) ? (g * 8u + 2u * e) : (16u + g * 8u + 2u * (e - 4u)));
    PackH p;
    p.h[0] = (_Float16)cb[n * 512u + k];
    p.h[1] = (_Float16)cb[n * 512u + k + 1u];
    B[t] = p.u;
}

// ---------------------------------------------------------------------------
// Kernel 3: GEMM [32768x512]x[512x1024] via v_wmma_f32_16x16x32_f16, fused
// per-row argmax. 8 waves/block share codebook B staged through a 2x64KB LDS
// double buffer (async global->LDS + s_wait_asynccnt when available).
// ---------------------------------------------------------------------------
__device__ __forceinline__ void gemm_chunk(const _Float16* bs, const v16h a[16],
                                           int chunk, int lane,
                                           float bestv[8], int besti[8]) {
    for (int ln = 0; ln < 4; ++ln) {        // 4 n-tiles per chunk
        v8f acc = {};
#pragma unroll
        for (int kt = 0; kt < 16; ++kt) {
            v16h bb = *(const v16h*)(bs + (((ln * 16 + kt) << 9) + (lane << 4)));
            acc = __builtin_amdgcn_wmma_f32_16x16x32_f16(
                false, a[kt], false, bb, (short)0, acc, false, false);
        }
        int ncol = (chunk * 4 + ln) * 16 + (lane & 15);
#pragma unroll
        for (int r = 0; r < 8; ++r) {
            float v = acc[r];
            if (v > bestv[r]) { bestv[r] = v; besti[r] = ncol; }  // strict >: first tie wins
        }
    }
}

__global__ void __launch_bounds__(256) vq_gemm_argmax(
    const _Float16* __restrict__ A, const _Float16* __restrict__ Bm,
    int* __restrict__ idx, int* __restrict__ counts) {
    __shared__ _Float16 Bs[2 * CH_HALVES];  // 128 KB double buffer
    const int tid  = threadIdx.x;
    const int lane = tid & 31;
    const int wave = tid >> 5;
    const int mt   = blockIdx.x * 8 + wave;

    // the wave's 16 A fragments (full K = 512) live in registers
    const v16h* Af = (const v16h*)A + (mt * 16) * 32 + lane;
    v16h a[16];
#pragma unroll
    for (int kt = 0; kt < 16; ++kt) a[kt] = Af[kt * 32];

    float bestv[8];
    int   besti[8];
#pragma unroll
    for (int r = 0; r < 8; ++r) { bestv[r] = -3.4e38f; besti[r] = 0; }

    copy_chunk(Bm, Bs, tid);                         // prime chunk 0
    for (int ch = 0; ch < 15; ++ch) {
        copy_chunk(Bm + (ch + 1) * CH_HALVES,        // stream next chunk
                   Bs + ((ch + 1) & 1) * CH_HALVES, tid);
        WAIT_ASYNC(16);                              // chunk ch resident (16 newest in flight)
        __syncthreads();
        gemm_chunk(Bs + (ch & 1) * CH_HALVES, a, ch, lane, bestv, besti);
        __syncthreads();
    }
    WAIT_ASYNC(0);
    __syncthreads();
    gemm_chunk(Bs + CH_HALVES, a, 15, lane, bestv, besti);

    // reduce across the 16 lanes sharing each row (xor<16 stays in-group)
#pragma unroll
    for (int off = 1; off < 16; off <<= 1) {
#pragma unroll
        for (int r = 0; r < 8; ++r) {
            float ov = __shfl_xor(bestv[r], off, 32);
            int   oi = __shfl_xor(besti[r], off, 32);
            if (ov > bestv[r] || (ov == bestv[r] && oi < besti[r])) {
                bestv[r] = ov; besti[r] = oi;
            }
        }
    }

    if ((lane & 15) == 0) {
        int mbase = mt * 16 + ((lane >> 4) << 3);    // lane0 -> rows 0..7, lane16 -> 8..15
#pragma unroll
        for (int r = 0; r < 8; ++r) {
            idx[mbase + r] = besti[r];
            atomicAdd(&counts[besti[r]], 1);
        }
    }
}

// ---------------------------------------------------------------------------
// Kernel 4: entropy / perplexity scalars
// ---------------------------------------------------------------------------
__global__ void vq_scalar(const int* __restrict__ counts, float* __restrict__ out) {
    __shared__ float red[1024];
    int t = threadIdx.x;
    float c = (float)counts[t];
    float p = c * (1.0f / 32768.0f);
    red[t] = (c > 0.0f) ? p * logf(p) : 0.0f;
    __syncthreads();
    for (int s = 512; s > 0; s >>= 1) {
        if (t < s) red[t] += red[t + s];
        __syncthreads();
    }
    if (t == 0) {
        float entropy = -red[0];
        float pl = expf(-entropy);   // 1 / exp(entropy)
        out[NQ]     = pl;            // perplexity_loss
        out[NQ + 1] = 0.25f * pl;    // loss = BETA * perplexity_loss
    }
}

// ---------------------------------------------------------------------------
// Kernel 5: straight-through output: out[b][c][w][h] = xs + (q - xs)
// ---------------------------------------------------------------------------
__global__ void __launch_bounds__(256) vq_gather(
    const float* __restrict__ x, const float* __restrict__ cb,
    const int* __restrict__ idx, float* __restrict__ out) {
    int bw = blockIdx.x;             // 0..1023
    int b  = bw >> 5, w = bw & 31;
    int h  = threadIdx.x & 31;
    int cs = threadIdx.x >> 5;       // 0..7
    int k  = idx[(b << 10) + (w << 5) + h];
    const float* crow = cb + k * 512;
    for (int c = cs; c < 512; c += 8) {
        int xi = (b * 512 + c) * 1024 + w * 32 + h;
        float xv = x[xi];
        float s  = 1.0f / (1.0f + expf(-100.0f * xv));
        float xs = s * 2.0f - 1.0f;
        float q  = crow[c];
        out[xi] = xs + (q - xs);     // same op order as reference STE
    }
}

extern "C" void kernel_launch(void* const* d_in, const int* in_sizes, int n_in,
                              void* d_out, int out_size, void* d_ws, size_t ws_size,
                              hipStream_t stream) {
    const float* x  = (const float*)d_in[0];   // [32,512,32,32]
    const float* cb = (const float*)d_in[1];   // [1024,512]
    float* out = (float*)d_out;
    char* ws = (char*)d_ws;

    // workspace layout
    _Float16* A   = (_Float16*)(ws);                         // 33,554,432 B
    _Float16* B   = (_Float16*)(ws + 33554432);              //  1,048,576 B
    int* idx      = (int*)(ws + 34603008);                   //    131,072 B
    int* counts   = (int*)(ws + 34734080);                   //      4,096 B

    vq_pack_a<<<2048, 256, 0, stream>>>(x, (unsigned*)A);
    vq_pack_b<<<1024, 256, 0, stream>>>(cb, (unsigned*)B);
    (void)hipMemsetAsync(counts, 0, K_CODES * sizeof(int), stream);
    vq_gemm_argmax<<<256, 256, 0, stream>>>(A, B, idx, counts);
    vq_scalar<<<1, 1024, 0, stream>>>(counts, out);
    vq_gather<<<1024, 256, 0, stream>>>(x, cb, idx, out);
}
